// PhaseMultiHeadAttention_14121852469302
// MI455X (gfx1250) — matrix operand order repaired and checked by reading, not compiled, not run on hardware
//
#include <hip/hip_runtime.h>
#include <math.h>

// ---------- types ----------
typedef __bf16 bf16_t;
typedef __attribute__((ext_vector_type(16))) bf16_t v16bf;
typedef __attribute__((ext_vector_type(8)))  bf16_t v8bf;
typedef __attribute__((ext_vector_type(8)))  float  v8f;
typedef __attribute__((ext_vector_type(2)))  float  v2f;

// ---------- problem constants ----------
#define B_  2
#define S_  2048
#define D_  512
#define H_  8
#define HD_ 64

#define TQ 128      // query rows per workgroup
#define TK 32       // key rows per k-iteration
#define QSTR 136    // LDS row stride (ushorts) for Q tile [TQ][2*HD]
#define KSTR 136    // LDS row stride for K tile  [TK][2*HD]
#define VSTR 40     // LDS row stride for V^T tile [2*HD][TK]
#define PSTR 40     // LDS row stride for P tile  [16][TK] per wave

__device__ __forceinline__ unsigned short f2bf(float f) {
    unsigned int u = __builtin_bit_cast(unsigned int, f);
    unsigned int r = u + 0x7FFFu + ((u >> 16) & 1u);   // round-to-nearest-even
    return (unsigned short)(r >> 16);
}

// DPP16 ROW_XOR_MASK exchange within 16-lane rows: pure VALU, no LDS traffic
template<int MASK>
__device__ __forceinline__ float rowxor(float x) {
    int xi = __builtin_bit_cast(int, x);
    int r  = __builtin_amdgcn_update_dpp(xi, xi, 0x160 | MASK, 0xF, 0xF, false);
    return __builtin_bit_cast(float, r);
}
__device__ __forceinline__ float rowmax16(float x) {
    x = fmaxf(x, rowxor<1>(x));
    x = fmaxf(x, rowxor<2>(x));
    x = fmaxf(x, rowxor<4>(x));
    x = fmaxf(x, rowxor<8>(x));
    return x;
}
__device__ __forceinline__ float rowsum16(float x) {
    x += rowxor<1>(x);
    x += rowxor<2>(x);
    x += rowxor<4>(x);
    x += rowxor<8>(x);
    return x;
}

// load a 16-element bf16 WMMA operand from two 16B LDS chunks
__device__ __forceinline__ v16bf ld_ab(const unsigned short* p0, const unsigned short* p1) {
    v8bf lo = *reinterpret_cast<const v8bf*>(p0);
    v8bf hi = *reinterpret_cast<const v8bf*>(p1);
    return __builtin_shufflevector(lo, hi, 0,1,2,3,4,5,6,7,8,9,10,11,12,13,14,15);
}

__global__ __launch_bounds__(256, 2)
void phase_attn_kernel(const float* __restrict__ sr, const float* __restrict__ si,
                       const float* __restrict__ qrot, const float* __restrict__ krot,
                       const float* __restrict__ vrot, float* __restrict__ out)
{
    __shared__ __align__(16) unsigned short q_s[TQ * QSTR];        // Q (qr|qi) bf16
    __shared__ __align__(16) unsigned short k_s[TK * KSTR];        // K (kr|ki) bf16
    __shared__ __align__(16) unsigned short v_s[(2*HD_) * VSTR];   // V^T (vr|vi) bf16
    __shared__ __align__(16) unsigned short p_s[8 * 16 * PSTR];    // per-wave P scratch

    const int blk  = blockIdx.x;
    const int qb   = blk % (S_/TQ);
    const int h    = (blk / (S_/TQ)) % H_;
    const int b    = blk / ((S_/TQ) * H_);
    const int qs   = qb * TQ;
    const int tid  = threadIdx.x;
    const int wave = tid >> 5;
    const int lane = tid & 31;
    const int half = lane >> 4;
    const int l16  = lane & 15;

    const size_t sbase = ((size_t)b * S_) * D_ + (size_t)h * HD_;

    // Each thread only ever fills one d-column (256 threads, 64-wide rows):
    // precompute all six rotation coefficients once, in registers.
    const int dfill = tid & (HD_-1);
    const int rfill = tid >> 6;            // base row within each fill pass
    float cq, sq, ck, sk, cv, sv;
    __sincosf(qrot[h*HD_ + dfill], &sq, &cq);
    __sincosf(krot[h*HD_ + dfill], &sk, &ck);
    __sincosf(vrot[h*HD_ + dfill], &sv, &cv);

    // ---- Stage Q block: rotate into bf16 LDS tile [m][0:64]=qr, [m][64:128]=qi ----
    #pragma unroll
    for (int i = 0; i < TQ/4; ++i) {
        int m = rfill + i*4;
        float ar = sr[sbase + (size_t)(qs+m)*D_ + dfill];
        float ai = si[sbase + (size_t)(qs+m)*D_ + dfill];
        q_s[m*QSTR + dfill]        = f2bf(ar*cq - ai*sq);
        q_s[m*QSTR + HD_ + dfill]  = f2bf(ar*sq + ai*cq);
    }
    __syncthreads();

    // ---- loop-invariant Q A-operands (16x128 split into 4 x K=32) ----
    v16bf a_q[4];
    #pragma unroll
    for (int st = 0; st < 4; ++st) {
        const unsigned short* p0 = &q_s[(wave*16 + l16)*QSTR + st*32 + half*8];
        a_q[st] = ld_ab(p0, p0 + 16);
    }

    // ---- flash state ----
    float Of[8][8];            // O accumulator: [d-tile][C-row]
    float mi[8], li[8];
    #pragma unroll
    for (int r = 0; r < 8; ++r) {
        mi[r] = -1e30f; li[r] = 0.f;
        #pragma unroll
        for (int d = 0; d < 8; ++d) Of[d][r] = 0.f;
    }

    const int row0 = qs + wave*16;
    const int nkb  = qs/TK + TQ/TK;

    for (int kb = 0; kb < nkb; ++kb) {
        const int t0 = kb * TK;

        __syncthreads();   // previous tile fully consumed
        // ---- stage K tile (row-major) and V tile (transposed); no sincos here ----
        #pragma unroll
        for (int i = 0; i < TK/4; ++i) {
            int t = rfill + i*4;
            float ar = sr[sbase + (size_t)(t0+t)*D_ + dfill];
            float ai = si[sbase + (size_t)(t0+t)*D_ + dfill];
            k_s[t*KSTR + dfill]          = f2bf(ar*ck - ai*sk);
            k_s[t*KSTR + HD_ + dfill]    = f2bf(ar*sk + ai*ck);
            v_s[dfill*VSTR + t]          = f2bf(ar*cv - ai*sv);
            v_s[(HD_+dfill)*VSTR + t]    = f2bf(ar*sv + ai*cv);
        }
        __syncthreads();

        // ---- S = Q K^T : two 16-col tiles, 4 WMMA steps each over K=128 ----
        float s0f[8], s1f[8];
        {
            v8f acc0, acc1;
            #pragma unroll
            for (int j = 0; j < 8; ++j) { acc0[j] = 0.f; acc1[j] = 0.f; }
            #pragma unroll
            for (int st = 0; st < 4; ++st) {
                const unsigned short* pb0 = &k_s[l16*KSTR + st*32 + half*8];
                v16bf b0 = ld_ab(pb0, pb0 + 16);
                acc0 = __builtin_amdgcn_wmma_f32_16x16x32_bf16(false, a_q[st], false, b0,
                                                               (short)0, acc0, false, false);
                const unsigned short* pb1 = &k_s[(16 + l16)*KSTR + st*32 + half*8];
                v16bf b1 = ld_ab(pb1, pb1 + 16);
                acc1 = __builtin_amdgcn_wmma_f32_16x16x32_bf16(false, a_q[st], false, b1,
                                                               (short)0, acc1, false, false);
            }
            #pragma unroll
            for (int j = 0; j < 8; ++j) { s0f[j] = acc0[j]; s1f[j] = acc1[j]; }
        }

        // ---- causal mask (only near/above the diagonal) ----
        if (t0 + TK - 1 > row0) {
            #pragma unroll
            for (int r = 0; r < 8; ++r) {
                int rg = row0 + r + half*8;
                if (t0 + l16      > rg) s0f[r] = -1e30f;
                if (t0 + 16 + l16 > rg) s1f[r] = -1e30f;
            }
        }

        // ---- online softmax (DPP row reductions within 16-lane half-groups) ----
        #pragma unroll
        for (int r = 0; r < 8; ++r) {
            float mx    = rowmax16(fmaxf(s0f[r], s1f[r]));
            float mnew  = fmaxf(mi[r], mx);
            float scale = __expf(mi[r] - mnew);
            float p0 = __expf(s0f[r] - mnew);
            float p1 = __expf(s1f[r] - mnew);
            float sum = rowsum16(p0 + p1);
            li[r] = li[r]*scale + sum;
            mi[r] = mnew;
            #pragma unroll
            for (int d = 0; d < 8; ++d) Of[d][r] *= scale;
            unsigned short* pp = &p_s[(wave*16 + r + half*8)*PSTR];
            pp[l16]      = f2bf(p0);
            pp[16 + l16] = f2bf(p1);
        }
        asm volatile("s_wait_dscnt 0" ::: "memory");  // P stores -> P reload (same wave)

        // ---- O += P (16x32) * V^T (32x128) : 8 N-tiles ----
        const unsigned short* pa = &p_s[(wave*16 + l16)*PSTR + half*8];
        v16bf a_p = ld_ab(pa, pa + 16);
        #pragma unroll
        for (int dt = 0; dt < 8; ++dt) {
            const unsigned short* pv = &v_s[(dt*16 + l16)*VSTR + half*8];
            v16bf b_v = ld_ab(pv, pv + 16);
            v8f c;
            #pragma unroll
            for (int j = 0; j < 8; ++j) c[j] = Of[dt][j];
            c = __builtin_amdgcn_wmma_f32_16x16x32_bf16(false, a_p, false, b_v,
                                                        (short)0, c, false, false);
            #pragma unroll
            for (int j = 0; j < 8; ++j) Of[dt][j] = c[j];
        }
    }

    // ---- epilogue: one v_rcp per row, paired {real,imag} b64 stores ----
    float inv[8];
    #pragma unroll
    for (int r = 0; r < 8; ++r) inv[r] = __builtin_amdgcn_rcpf(li[r]);

    #pragma unroll
    for (int dt = 0; dt < 4; ++dt) {
        int d = dt*16 + l16;                       // real tile dt, imag tile dt+4
        #pragma unroll
        for (int r = 0; r < 8; ++r) {
            int sg = row0 + r + half*8;
            v2f val;
            val[0] = Of[dt][r]     * inv[r];       // real
            val[1] = Of[dt + 4][r] * inv[r];       // imag
            *reinterpret_cast<v2f*>(&out[(((size_t)b*S_ + sg)*D_ + h*HD_ + d)*2]) = val;
        }
    }
}

extern "C" void kernel_launch(void* const* d_in, const int* in_sizes, int n_in,
                              void* d_out, int out_size, void* d_ws, size_t ws_size,
                              hipStream_t stream) {
    (void)in_sizes; (void)n_in; (void)d_ws; (void)ws_size; (void)out_size;
    const float* sr = (const float*)d_in[0];
    const float* si = (const float*)d_in[1];
    const float* qr = (const float*)d_in[2];
    const float* kr = (const float*)d_in[3];
    const float* vr = (const float*)d_in[4];
    float* out = (float*)d_out;
    dim3 grid(B_ * H_ * (S_ / TQ));
    dim3 block(256);
    hipLaunchKernelGGL(phase_attn_kernel, grid, block, 0, stream, sr, si, qr, kr, vr, out);
}
